// EdgeConvLayer_79216376807661
// MI455X (gfx1250) — compile-verified, matrix operand order checked by reading, not run on hardware
//
#include <hip/hip_runtime.h>
#include <hip/hip_bf16.h>

typedef __attribute__((ext_vector_type(16))) __bf16 v16bf;
typedef __attribute__((ext_vector_type(8)))  float  v8f;
typedef __attribute__((ext_vector_type(4)))  float  v4f;

#define WAVES_PER_BLOCK 8
#define STG_STRIDE 68          // 16x64 f32 staging, padded to avoid bank conflicts
#define NFRAG 24               // 16 frags for W1 (4 K-chunks x 4 N-tiles), 8 for W2

__global__ void zero_f32(float* __restrict__ p, int n) {
    int i = blockIdx.x * blockDim.x + threadIdx.x;
    if (i < n) p[i] = 0.f;
}

__global__ __launch_bounds__(256)
void edgeconv_wmma_kernel(const float* __restrict__ h, const float* __restrict__ e,
                          const int* __restrict__ src, const int* __restrict__ dst,
                          const float* __restrict__ W1, const float* __restrict__ b1,
                          const float* __restrict__ W2, const float* __restrict__ b2,
                          float* __restrict__ out, int nEdges)
{
    // bf16 B-fragments, laid out exactly as the WMMA VGPR striping expects:
    // fragment f, lane l -> 16 bf16 (32B), loaded back as one v16bf.
    __shared__ __bf16 wfrag[NFRAG * 512];
    __shared__ float  staging[WAVES_PER_BLOCK][16 * STG_STRIDE];

    const int tid  = threadIdx.x;
    const int lane = tid & 31;
    const int wave = tid >> 5;
    const int half = lane >> 4;     // 0: lanes 0-15, 1: lanes 16-31
    const int col  = lane & 15;

    // ---- pack W1/W2 (fp32 row-major) into bf16 WMMA B-fragments in LDS ----
    for (int p = tid; p < NFRAG * 32; p += blockDim.x) {
        int f = p >> 5;
        int l = p & 31;
        int lh = l >> 4, lc = l & 15;
        const float* W;
        int chunk, ntile;
        if (f < 16) { chunk = f >> 2;        ntile = f & 3;          W = W1; }
        else        { int g = f - 16; chunk = g >> 2; ntile = g & 3; W = W2; }
        int N = ntile * 16 + lc;
        __bf16* q = &wfrag[f * 512 + l * 16];
#pragma unroll
        for (int j = 0; j < 16; ++j) {
            // 16-bit A/B striping: elems 0..7 -> K = 8*half + 0..7,
            //                      elems 8..15 -> K = 16 + 8*half + 0..7
            int kl = (j < 8) ? (8 * lh + j) : (16 + 8 * lh + (j - 8));
            int K  = chunk * 32 + kl;
            q[j] = (__bf16)W[K * 64 + N];
        }
    }
    __syncthreads();

    // W2 B-fragments are small (8 x 8 VGPRs): keep them in registers for the
    // whole tile loop instead of re-reading LDS every tile.
    v16bf w2f[8];
#pragma unroll
    for (int i = 0; i < 8; ++i)
        w2f[i] = *(const v16bf*)&wfrag[(16 + i) * 512 + lane * 16];

    // per-lane hoisted constants (N = t*16 + col)
    float b1v[4], b2v[4], w1e[4];
#pragma unroll
    for (int t = 0; t < 4; ++t) {
        b1v[t] = b1[t * 16 + col];
        b2v[t] = b2[t * 16 + col];
        w1e[t] = W1[128 * 64 + t * 16 + col];   // K=128 row (edge-feature input)
    }

    float* stg = staging[wave];
    const int waveId   = blockIdx.x * WAVES_PER_BLOCK + wave;
    const int numWaves = gridDim.x * WAVES_PER_BLOCK;
    const int numTiles = (nEdges + 15) >> 4;

    for (int tile = waveId; tile < numTiles; tile += numWaves) {
        int eidx = tile * 16 + col;                    // this lane's edge row
        int ecl  = eidx < nEdges ? eidx : (nEdges - 1);
        int s    = src[ecl];
        int d    = dst[ecl];
        float ev = e[ecl];
        const float* hs = h + (size_t)s * 64;
        const float* hd = h + (size_t)d * 64;

        // ---- layer 1: C1 = [h_src|h_dst](16x128) @ W1 + b1 + e * W1[128,:] ----
        v8f c1[4];
#pragma unroll
        for (int t = 0; t < 4; ++t) {
#pragma unroll
            for (int v = 0; v < 8; ++v) {
                int M = v + 8 * half;
                float em = __shfl(ev, M, 32);
                c1[t][v] = b1v[t] + em * w1e[t];
            }
        }

#pragma unroll
        for (int c = 0; c < 4; ++c) {
            const float* rp = (c < 2) ? hs : hd;       // K 0..63 = h_src, 64..127 = h_dst
            int base = (c & 1) * 32 + 8 * half;
            v4f x0 = *(const v4f*)(rp + base);
            v4f x1 = *(const v4f*)(rp + base + 4);
            v4f x2 = *(const v4f*)(rp + base + 16);
            v4f x3 = *(const v4f*)(rp + base + 20);
            v16bf a;
#pragma unroll
            for (int j = 0; j < 4; ++j) {
                a[j]      = (__bf16)x0[j];
                a[4 + j]  = (__bf16)x1[j];
                a[8 + j]  = (__bf16)x2[j];
                a[12 + j] = (__bf16)x3[j];
            }
#pragma unroll
            for (int t = 0; t < 4; ++t) {
                v16bf b = *(const v16bf*)&wfrag[(c * 4 + t) * 512 + lane * 16];
                c1[t] = __builtin_amdgcn_wmma_f32_16x16x32_bf16(
                            false, a, false, b, (short)0, c1[t], false, false);
            }
        }

        // ---- relu + cross-lane transpose (C layout -> A layout) via LDS ----
        // Same-wave LDS ops are in-order: no barrier needed.
#pragma unroll
        for (int t = 0; t < 4; ++t) {
#pragma unroll
            for (int v = 0; v < 8; ++v) {
                int M = v + 8 * half;
                float x = c1[t][v];
                stg[M * STG_STRIDE + t * 16 + col] = x > 0.f ? x : 0.f;
            }
        }

        // ---- layer 2: C2 = relu(C1)(16x64) @ W2 + b2 (weights in registers) ----
        v8f c2[4];
#pragma unroll
        for (int t = 0; t < 4; ++t)
#pragma unroll
            for (int v = 0; v < 8; ++v) c2[t][v] = b2v[t];

#pragma unroll
        for (int c = 0; c < 2; ++c) {
            int base = c * 32 + 8 * half;
            const float* rp = stg + col * STG_STRIDE;  // A row M = col
            v4f x0 = *(const v4f*)(rp + base);
            v4f x1 = *(const v4f*)(rp + base + 4);
            v4f x2 = *(const v4f*)(rp + base + 16);
            v4f x3 = *(const v4f*)(rp + base + 20);
            v16bf a;
#pragma unroll
            for (int j = 0; j < 4; ++j) {
                a[j]      = (__bf16)x0[j];
                a[4 + j]  = (__bf16)x1[j];
                a[8 + j]  = (__bf16)x2[j];
                a[12 + j] = (__bf16)x3[j];
            }
#pragma unroll
            for (int t = 0; t < 4; ++t) {
                c2[t] = __builtin_amdgcn_wmma_f32_16x16x32_bf16(
                            false, a, false, w2f[c * 4 + t], (short)0, c2[t], false, false);
            }
        }

        // ---- scatter-add messages into out[dst] ----
#pragma unroll
        for (int v = 0; v < 8; ++v) {
            int M  = v + 8 * half;
            int dn = __shfl(d, M, 32);                 // dst of edge row M
            int gi = tile * 16 + M;
            if (gi < nEdges) {
                float* op = out + (size_t)dn * 64;
#pragma unroll
                for (int t = 0; t < 4; ++t)
                    unsafeAtomicAdd(op + t * 16 + col, c2[t][v]);
            }
        }
    }
}

extern "C" void kernel_launch(void* const* d_in, const int* in_sizes, int n_in,
                              void* d_out, int out_size, void* d_ws, size_t ws_size,
                              hipStream_t stream) {
    const float* h   = (const float*)d_in[0];
    const float* e   = (const float*)d_in[1];
    const int*   src = (const int*)d_in[2];
    const int*   dst = (const int*)d_in[3];
    const float* W1  = (const float*)d_in[4];
    const float* b1  = (const float*)d_in[5];
    const float* W2  = (const float*)d_in[6];
    const float* b2  = (const float*)d_in[7];
    float* out = (float*)d_out;
    int nEdges = in_sizes[2];

    // output is accumulated via atomics -> must start from zero every call
    zero_f32<<<(out_size + 255) / 256, 256, 0, stream>>>(out, out_size);

    int numTiles = (nEdges + 15) / 16;
    int blocks = (numTiles + WAVES_PER_BLOCK - 1) / WAVES_PER_BLOCK;
    if (blocks > 2048) blocks = 2048;   // waves grid-stride over tiles
    edgeconv_wmma_kernel<<<blocks, 256, 0, stream>>>(h, e, src, dst, W1, b1, W2, b2,
                                                     out, nEdges);
}
